// GDU_30425548325390
// MI455X (gfx1250) — matrix operand order repaired and checked by reading, not compiled, not run on hardware
//
#include <hip/hip_runtime.h>
#include <math.h>

// ---------------------------------------------------------------------------
// GDU cell, MI455X (gfx1250, wave32, WMMA). Two fused bf16-WMMA passes.
// ---------------------------------------------------------------------------

typedef __bf16 bf16;
typedef bf16  v16bf __attribute__((ext_vector_type(16)));
typedef bf16  v8bf  __attribute__((ext_vector_type(8)));
typedef bf16  v4bf  __attribute__((ext_vector_type(4)));
typedef float v8f   __attribute__((ext_vector_type(8)));

#define NROWS   65536
#define DIM     256
#define TM      32          // rows per workgroup tile
#define LDSTR   264         // DIM + 8 bf16 padding (bank spread)
#define THREADS 256         // 8 waves: 2 row-slabs x 4 col-slabs

__device__ __forceinline__ bf16 to_bf16(float f) {
  union { float f; unsigned u; } v; v.f = f;
  unsigned r = v.u + 0x7FFFu + ((v.u >> 16) & 1u);   // round-to-nearest-even
  unsigned short hs = (unsigned short)(r >> 16);
  bf16 b; __builtin_memcpy(&b, &hs, 2);
  return b;
}

__device__ __forceinline__ float sigmoidf_(float x) {
  return 1.0f / (1.0f + __expf(-x));
}

// A-fragment for v_wmma_f32_16x16x32_bf16 from a row-major bf16 LDS tile.
// lane = m + 16*s holds K = {k0+8s..k0+8s+7} and {k0+16+8s..k0+16+8s+7}.
__device__ __forceinline__ v16bf load_a_frag(const bf16* __restrict__ lrow,
                                             int k0, int sshift) {
  v8bf lo = *(const v8bf*)(lrow + k0 + sshift);
  v8bf hi = *(const v8bf*)(lrow + k0 + 16 + sshift);
  return __builtin_shufflevector(lo, hi,
      0,1,2,3,4,5,6,7,8,9,10,11,12,13,14,15);
}

// ---------------------------------------------------------------------------
// Weight fp32 -> bf16 conversion (19 matrices of 256x256 into workspace)
// ---------------------------------------------------------------------------
struct WPtrs { const float* p[19]; };

__global__ __launch_bounds__(256)
void convert_w(WPtrs w, bf16* __restrict__ dst) {
  int mat = blockIdx.y;
  int i = blockIdx.x * 256 + threadIdx.x;                  // 0..65535
  dst[(size_t)mat * (DIM * DIM) + i] = to_bf16(w.p[mat][i]);
}

// ---------------------------------------------------------------------------
// Phase 1: f = sig(x Wfx + z Wfz + h Wfh + bf); e likewise.
// Writes z_tilde = f*z and h_tilde = e*h as bf16 into workspace.
// ---------------------------------------------------------------------------
__global__ __launch_bounds__(THREADS)
void gdu_phase1(const float* __restrict__ x, const float* __restrict__ z,
                const float* __restrict__ h, const bf16* __restrict__ wB,
                const float* __restrict__ b_f, const float* __restrict__ b_e,
                bf16* __restrict__ zt, bf16* __restrict__ ht) {
  __shared__ bf16 lds[TM * LDSTR];

  const int tid = threadIdx.x;
  const int lane = tid & 31, wid = tid >> 5;
  const int rowSlab = wid & 1;         // 2 x 16 rows
  const int colSlab = wid >> 1;        // 4 x 64 cols
  const int row0 = blockIdx.x * TM;
  const int colBase = colSlab * 64;
  const int sshift = (lane >> 4) * 8;
  const int mrow = rowSlab * 16 + (lane & 15);

  v8f acc[2][4] = {};                  // [gate f|e][4 col tiles]

  const float* srcs[3] = { x, z, h };
  constexpr int WSLOT[3][2] = { {0,3}, {1,4}, {2,5} };  // {Wf*, We*} per src

  #pragma unroll
  for (int src = 0; src < 3; ++src) {
    __syncthreads();
    { // stage fp32 tile -> bf16 LDS (32 x 256, each thread 8 x float4)
      const float* S = srcs[src] + (size_t)row0 * DIM;
      #pragma unroll
      for (int i = 0; i < 8; ++i) {
        int e = tid + i * THREADS;
        int idx = e * 4, rr = idx >> 8, cc = idx & 255;
        float4 v = *(const float4*)(S + idx);
        v4bf o; o[0]=to_bf16(v.x); o[1]=to_bf16(v.y);
                o[2]=to_bf16(v.z); o[3]=to_bf16(v.w);
        *(v4bf*)&lds[rr * LDSTR + cc] = o;
      }
    }
    __syncthreads();

    const bf16* lrow = &lds[mrow * LDSTR];
    for (int k0 = 0; k0 < DIM; k0 += 32) {
      v16bf a = load_a_frag(lrow, k0, sshift);
      #pragma unroll
      for (int g = 0; g < 2; ++g) {
        const bf16* Wp = wB + (size_t)WSLOT[src][g] * (DIM * DIM)
                            + (size_t)k0 * DIM + colBase + (lane & 15);
        #pragma unroll
        for (int t = 0; t < 4; ++t) {
          v16bf b;
          #pragma unroll
          for (int q = 0; q < 16; ++q)
            b[q] = Wp[(size_t)(q < 8 ? sshift + q : 16 + sshift + q - 8) * DIM + t * 16];
          acc[g][t] = __builtin_amdgcn_wmma_f32_16x16x32_bf16(
              false, a, false, b, (short)0, acc[g][t], false, false);
        }
      }
    }
  }

  const int s = lane >> 4, nl = lane & 15;
  #pragma unroll
  for (int t = 0; t < 4; ++t) {
    int col = colBase + t * 16 + nl;
    float bfv = b_f[col], bev = b_e[col];
    #pragma unroll
    for (int j = 0; j < 8; ++j) {
      int row = row0 + rowSlab * 16 + j + 8 * s;
      size_t off = (size_t)row * DIM + col;
      float fv = sigmoidf_(acc[0][t][j] + bfv);
      float ev = sigmoidf_(acc[1][t][j] + bev);
      zt[off] = to_bf16(fv * z[off]);
      ht[off] = to_bf16(ev * h[off]);
    }
  }
}

// ---------------------------------------------------------------------------
// Phase 2: g_pre, r_pre (5-src GEMM sums) + ux, uz_z, uz_zt, uh_h, uh_ht,
// then fused sigmoid/tanh blend epilogue -> fp32 out.
// ---------------------------------------------------------------------------
__global__ __launch_bounds__(THREADS)
void gdu_phase2(const float* __restrict__ x, const float* __restrict__ z,
                const float* __restrict__ h, const bf16* __restrict__ wB,
                const bf16* __restrict__ zt, const bf16* __restrict__ ht,
                const float* __restrict__ b_g, const float* __restrict__ b_r,
                const float* __restrict__ b_u, float* __restrict__ out) {
  __shared__ bf16 lds[TM * LDSTR];

  const int tid = threadIdx.x;
  const int lane = tid & 31, wid = tid >> 5;
  const int rowSlab = wid & 1;
  const int colSlab = wid >> 1;
  const int row0 = blockIdx.x * TM;
  const int colBase = colSlab * 64;
  const int sshift = (lane >> 4) * 8;
  const int mrow = rowSlab * 16 + (lane & 15);

  // acc index: 0=g_pre 1=r_pre 2=ux 3=uz_z 4=uz_zt 5=uh_h 6=uh_ht
  v8f acc[7][4] = {};

  const float* f32src[3] = { x, z, h };
  // sources: 0=x 1=z 2=h 3=z_tilde 4=h_tilde
  constexpr int WSLOT[5][3] = { {6,11,16}, {7,12,17}, {8,13,18},
                                {9,14,17}, {10,15,18} };
  constexpr int WACC[5][3]  = { {0,1,2}, {0,1,3}, {0,1,5},
                                {0,1,4}, {0,1,6} };

  #pragma unroll
  for (int src = 0; src < 5; ++src) {
    __syncthreads();
    if (src < 3) {            // fp32 -> bf16 stage
      const float* S = f32src[src] + (size_t)row0 * DIM;
      #pragma unroll
      for (int i = 0; i < 8; ++i) {
        int e = tid + i * THREADS;
        int idx = e * 4, rr = idx >> 8, cc = idx & 255;
        float4 v = *(const float4*)(S + idx);
        v4bf o; o[0]=to_bf16(v.x); o[1]=to_bf16(v.y);
                o[2]=to_bf16(v.z); o[3]=to_bf16(v.w);
        *(v4bf*)&lds[rr * LDSTR + cc] = o;
      }
    } else {                  // bf16 copy from workspace
      const bf16* S = (src == 3 ? zt : ht) + (size_t)row0 * DIM;
      #pragma unroll
      for (int i = 0; i < 8; ++i) {
        int e = tid + i * THREADS;
        int idx = e * 4, rr = idx >> 8, cc = idx & 255;
        *(v4bf*)&lds[rr * LDSTR + cc] = *(const v4bf*)(S + idx);
      }
    }
    __syncthreads();

    const bf16* lrow = &lds[mrow * LDSTR];
    for (int k0 = 0; k0 < DIM; k0 += 32) {
      v16bf a = load_a_frag(lrow, k0, sshift);
      #pragma unroll
      for (int wi = 0; wi < 3; ++wi) {
        const bf16* Wp = wB + (size_t)WSLOT[src][wi] * (DIM * DIM)
                            + (size_t)k0 * DIM + colBase + (lane & 15);
        #pragma unroll
        for (int t = 0; t < 4; ++t) {
          v16bf b;
          #pragma unroll
          for (int q = 0; q < 16; ++q)
            b[q] = Wp[(size_t)(q < 8 ? sshift + q : 16 + sshift + q - 8) * DIM + t * 16];
          acc[WACC[src][wi]][t] = __builtin_amdgcn_wmma_f32_16x16x32_bf16(
              false, a, false, b, (short)0, acc[WACC[src][wi]][t],
              false, false);
        }
      }
    }
  }

  const int s = lane >> 4, nl = lane & 15;
  #pragma unroll
  for (int t = 0; t < 4; ++t) {
    int col = colBase + t * 16 + nl;
    float bgv = b_g[col], brv = b_r[col], buv = b_u[col];
    #pragma unroll
    for (int j = 0; j < 8; ++j) {
      int row = row0 + rowSlab * 16 + j + 8 * s;
      float g  = sigmoidf_(acc[0][t][j] + bgv);
      float r  = sigmoidf_(acc[1][t][j] + brv);
      float ux = acc[2][t][j] + buv;
      float uzz = acc[3][t][j], uzzt = acc[4][t][j];
      float uhh = acc[5][t][j], uhht = acc[6][t][j];
      float o1 = tanhf(ux + uzzt + uhht);
      float o2 = tanhf(ux + uzz  + uhht);
      float o3 = tanhf(ux + uzzt + uhh );
      float o4 = tanhf(ux + uzz  + uhh );
      out[(size_t)row * DIM + col] =
          g * r * o1 + (1.0f - g) * r * o2 +
          g * (1.0f - r) * o3 + (1.0f - g) * (1.0f - r) * o4;
    }
  }
}

// ---------------------------------------------------------------------------
extern "C" void kernel_launch(void* const* d_in, const int* in_sizes, int n_in,
                              void* d_out, int out_size, void* d_ws,
                              size_t ws_size, hipStream_t stream) {
  const float* x   = (const float*)d_in[0];
  const float* z   = (const float*)d_in[1];
  const float* h   = (const float*)d_in[2];
  const float* b_f = (const float*)d_in[6];
  const float* b_e = (const float*)d_in[10];
  const float* b_g = (const float*)d_in[16];
  const float* b_r = (const float*)d_in[22];
  const float* b_u = (const float*)d_in[26];

  // bf16 weight slot order:
  // 0:Wfx 1:Wfz 2:Wfh | 3:Wex 4:Wez 5:Weh | 6:Wgx 7:Wgz 8:Wgh 9:Wgzt 10:Wght
  // 11:Wrx 12:Wrz 13:Wrh 14:Wrzt 15:Wrht | 16:Wux 17:Wuz 18:Wuh
  const int wIdx[19] = { 3,4,5, 7,8,9, 11,12,13,14,15, 17,18,19,20,21,
                         23,24,25 };
  WPtrs wp;
  for (int i = 0; i < 19; ++i) wp.p[i] = (const float*)d_in[wIdx[i]];

  bf16* wB = (bf16*)d_ws;
  size_t wBytes = (size_t)19 * DIM * DIM * sizeof(bf16);   // ~2.4 MB
  bf16* ztw = (bf16*)((char*)d_ws + wBytes);               // 32 MB
  bf16* htw = ztw + (size_t)NROWS * DIM;                   // 32 MB

  convert_w<<<dim3(DIM * DIM / 256, 19), 256, 0, stream>>>(wp, wB);
  gdu_phase1<<<NROWS / TM, THREADS, 0, stream>>>(x, z, h, wB, b_f, b_e,
                                                 ztw, htw);
  gdu_phase2<<<NROWS / TM, THREADS, 0, stream>>>(x, z, h, wB, ztw, htw,
                                                 b_g, b_r, b_u,
                                                 (float*)d_out);
}
